// SphericalConvolution_31439160607220
// MI455X (gfx1250) — matrix operand order repaired and checked by reading, not compiled, not run on hardware
//
#include <hip/hip_runtime.h>
#include <math.h>

typedef __attribute__((ext_vector_type(16))) _Float16 v16h;
typedef __attribute__((ext_vector_type(8)))  float    v8f;

#define NNODES    1024
#define CIN       32
#define COUT      32
#define KSH       25
#define RBETA     64
#define RALPHA    128
#define NBA       (RBETA * RALPHA)      // 8192
#define NROWS     (NNODES * CIN)        // 32768
#define NTILES    (NROWS / 16)          // 2048
#define NCHUNK    (NBA / 32)            // 256
// 1/sqrt(E[tanh(N(0,1))^2])  (reference estimates it from a fixed RNG sample)
#define ACT_CONST 1.5926100f

// ---- workspace byte offsets (all 256-aligned) ----
#define WS_Y      0u                         // 64 f32
#define WS_QW     256u                       // 64 f32
#define WS_PN     512u                       // 64*25 f32 (6400B)
#define WS_BANK1  8192u                      // [512 tiles][32 lanes][16] f16 = 512KB
#define WS_PROJ   (8192u + 524288u)          // [256 chunks][2 tiles][32 lanes][16] f16 = 512KB
#define WS_WL     (8192u + 1048576u)         // 32*32*5 f32 (20480B)
#define WS_XP     (8192u + 1048576u + 32768u)// 32768*32 f32 = 4MB

__constant__ int c_lofk[25] = {0, 1,1,1, 2,2,2,2,2, 3,3,3,3,3,3,3, 4,4,4,4,4,4,4,4,4};

// Branchless tanh: hardware V_TANH_F32 when the toolchain exposes it, else
// tanh(x) = 1 - 2/(exp(2x)+1) via hardware exp2/rcp (no EXEC divergence).
__device__ inline float fast_tanh(float v) {
#if __has_builtin(__builtin_amdgcn_tanhf)
  return __builtin_amdgcn_tanhf(v);
#elif __has_builtin(__builtin_amdgcn_tanh_f32)
  return __builtin_amdgcn_tanh_f32(v);
#else
  float e = __builtin_amdgcn_exp2f(v * 2.88539008177793f);   // 2*log2(e)
  return 1.0f - 2.0f * __builtin_amdgcn_rcpf(e + 1.0f);
#endif
}

// ---------------- setup: Gauss-Legendre nodes/weights + normalized assoc. Legendre --------
__global__ void gl_setup_kernel(float* __restrict__ ynode, float* __restrict__ qw,
                                float* __restrict__ Pn) {
  int i = threadIdx.x;
  if (i >= RBETA) return;
  const int n = RBETA;
  int j = n - 1 - i;                                  // ascending node order (numpy leggauss)
  double x = cos(M_PI * (j + 0.75) / (n + 0.5));
  double dp = 1.0;
  for (int it = 0; it < 64; ++it) {
    double p0 = 1.0, p1 = x;
    for (int k = 2; k <= n; ++k) {
      double p2 = ((2.0 * k - 1.0) * x * p1 - (k - 1.0) * p0) / (double)k;
      p0 = p1; p1 = p2;
    }
    dp = n * (x * p1 - p0) / (x * x - 1.0);
    x -= p1 / dp;
  }
  { // final derivative at converged root for the weight
    double p0 = 1.0, p1 = x;
    for (int k = 2; k <= n; ++k) {
      double p2 = ((2.0 * k - 1.0) * x * p1 - (k - 1.0) * p0) / (double)k;
      p0 = p1; p1 = p2;
    }
    dp = n * (x * p1 - p0) / (x * x - 1.0);
  }
  ynode[i] = (float)x;
  qw[i]    = (float)(2.0 / ((1.0 - x * x) * dp * dp));

  double P[5][5] = {};
  double y = x, s = sqrt(fmax(1.0 - y * y, 0.0));
  P[0][0] = 1.0;
  for (int m = 1; m <= 4; ++m) P[m][m] = -(2.0 * m - 1.0) * s * P[m - 1][m - 1];
  for (int m = 0; m < 4; ++m)  P[m + 1][m] = (2.0 * m + 1.0) * y * P[m][m];
  for (int m = 0; m <= 4; ++m)
    for (int l = m + 2; l <= 4; ++l)
      P[l][m] = ((2.0 * l - 1.0) * y * P[l - 1][m] - (l + m - 1.0) * P[l - 2][m]) / (double)(l - m);
  for (int l = 0; l <= 4; ++l)
    for (int m = 0; m <= l; ++m) {
      double ratio = 1.0;                              // (l-m)!/(l+m)!
      for (int t = l - m + 1; t <= l + m; ++t) ratio /= (double)t;
      double nlm = sqrt((2.0 * l + 1.0) / (4.0 * M_PI) * ratio);
      Pn[i * 25 + l * 5 + m] = (float)(P[l][m] * nlm);
    }
}

__device__ inline float sh_val(const float* __restrict__ Pn, int k, int ba) {
  int l = c_lofk[k];
  int m = k - l * l - l;
  int beta = ba >> 7, alpha = ba & 127;
  float p = Pn[beta * 25 + l * 5 + (m < 0 ? -m : m)];
  float ang = (float)(2.0 * M_PI) * (float)alpha * (1.0f / 128.0f);
  float f;
  if (m > 0)       f = 1.41421356237f * cosf((float)m * ang);
  else if (m == 0) f = 1.0f;
  else             f = 1.41421356237f * sinf((float)(-m) * ang);
  return p * f;
}

// B-operand bank for GEMM1, pre-swizzled: [tile t][lane: n=lane%16][halfword: K table order] (B layout: lane=N, regs=K)
__global__ void fill_bank1_kernel(const float* __restrict__ Pn, _Float16* __restrict__ bank1) {
  int tid = blockIdx.x * blockDim.x + threadIdx.x;    // 512*32*16 = 262144
  if (tid >= 512 * 32 * 16) return;
  int nn = tid & 15;
  int l  = (tid >> 4) & 31;
  int t  = tid >> 9;                                  // ba tile
  int hk = l >> 4;
  int K  = (nn < 8) ? (8 * hk + nn) : (16 + 8 * hk + (nn - 8));
  int ba = t * 16 + (l & 15);
  float v = (K < KSH) ? sh_val(Pn, K, ba) : 0.0f;
  bank1[tid] = (_Float16)v;
}

// A-operand bank for GEMM2 (proj^T tiles * ACT_CONST folded in), 16-bit A 16x32 layout
__global__ void fill_proj_kernel(const float* __restrict__ Pn, const float* __restrict__ qw,
                                 _Float16* __restrict__ projbank) {
  int tid = blockIdx.x * blockDim.x + threadIdx.x;    // 256*2*32*16 = 262144
  if (tid >= 256 * 2 * 32 * 16) return;
  int i  = tid & 15;
  int l  = (tid >> 4) & 31;
  int t  = (tid >> 9) & 1;
  int c  = tid >> 10;
  int kp = 16 * t + (l & 15);                         // M = k'
  int hh = l >> 4;
  int K  = (i < 8) ? (8 * hh + i) : (16 + 8 * hh + (i - 8));  // ISA A-layout K order
  int ba = 32 * c + K;
  float v = 0.0f;
  if (kp < KSH) {
    int beta = ba >> 7;
    v = sh_val(Pn, kp, ba) * qw[beta] * (float)(2.0 * M_PI / 128.0) * ACT_CONST;
  }
  projbank[tid] = (_Float16)v;
}

// w_l[j][i][l] = 2*pi*sqrt(4*pi/(2l+1)) * sum_b h[j,i,b] * Pn[b,l,0] * qw[b]
__global__ void fill_w_kernel(const float* __restrict__ h, const float* __restrict__ Pn,
                              const float* __restrict__ qw, float* __restrict__ wl) {
  int tid = blockIdx.x * blockDim.x + threadIdx.x;    // 32*32*5 = 5120
  if (tid >= COUT * CIN * 5) return;
  int l  = tid % 5;
  int ji = tid / 5;
  float acc = 0.0f;
  for (int b = 0; b < RBETA; ++b)
    acc += h[ji * RBETA + b] * Pn[b * 25 + l * 5] * qw[b];
  float scale = (float)(2.0 * M_PI) * sqrtf((float)(4.0 * M_PI) / (2.0f * (float)l + 1.0f));
  wl[tid] = scale * acc;
}

// ---------------- fused main kernel: x -> grid -> tanh -> x'  (all WMMA) ----------------
__global__ __launch_bounds__(256) void sph_main_kernel(
    const float* __restrict__ x, const _Float16* __restrict__ bank1,
    const _Float16* __restrict__ projbank, float* __restrict__ xprime) {
  __shared__ _Float16 lds[8][512];                    // 1KB per wave: grid^T chunk (32 ba x 16 nc)
  const int lane = threadIdx.x & 31;
  const int wave = threadIdx.x >> 5;
  const int tile = blockIdx.x * 8 + wave;             // grid sized exactly: no bounds check, EXEC all 1s
  const int rowbase = tile << 4;
  const int hh = lane >> 4;
  const int mm = lane & 15;

  // A1 = x tile (16 rows x 32 k, k>=25 padded 0) in 16-bit A layout
  const float* xrow = x + (size_t)(rowbase + mm) * KSH;
  v16h a1;
#pragma unroll
  for (int i = 0; i < 16; ++i) {
    int K = (i < 8) ? (8 * hh + i) : (16 + 8 * hh + (i - 8));
    float v = (K < KSH) ? xrow[K] : 0.0f;
    a1[i] = (_Float16)v;
  }

  v8f acc0 = {0.f,0.f,0.f,0.f,0.f,0.f,0.f,0.f};
  v8f acc1 = {0.f,0.f,0.f,0.f,0.f,0.f,0.f,0.f};
  _Float16* ldsw = &lds[wave][0];
  unsigned int* ldsw32 = (unsigned int*)ldsw;

  for (int c = 0; c < NCHUNK; ++c) {
    // GEMM1: grid(16 nc x 32 ba) = x_tile @ B  (two 16x16 WMMAs)
    const v16h b0 = *(const v16h*)(bank1 + ((size_t)(2 * c)     * 32 + lane) * 16);
    const v16h b1 = *(const v16h*)(bank1 + ((size_t)(2 * c + 1) * 32 + lane) * 16);
    v8f z = {0.f,0.f,0.f,0.f,0.f,0.f,0.f,0.f};
    v8f d0 = __builtin_amdgcn_wmma_f32_16x16x32_f16(false, a1, false, b0, (short)0, z, false, false);
    v8f d1 = __builtin_amdgcn_wmma_f32_16x16x32_f16(false, a1, false, b1, (short)0, z, false, false);

    // branchless activation + lane-transpose into LDS: grid^T[ba_local][nc] as f16
#pragma unroll
    for (int r = 0; r < 8; r += 2) {
      int nc = r + 8 * hh;                            // even; pack nc,nc+1 into one dword
      union { _Float16 hf[2]; unsigned int u; } p0, p1;
      p0.hf[0] = (_Float16)fast_tanh(d0[r]);
      p0.hf[1] = (_Float16)fast_tanh(d0[r + 1]);
      p1.hf[0] = (_Float16)fast_tanh(d1[r]);
      p1.hf[1] = (_Float16)fast_tanh(d1[r + 1]);
      ldsw32[(mm * 16 + nc) >> 1]          = p0.u;    // ba_local = mm      (tile 0)
      ldsw32[(((16 + mm) * 16) + nc) >> 1] = p1.u;    // ba_local = 16 + mm (tile 1)
    }
    __asm__ volatile("s_wait_dscnt 0x0" ::: "memory"); // CDNA5 split counter: order DS store->load
    v16h bg;                                           // B layout: lane = nc, regs = K(=ba_local) table order
#pragma unroll
    for (int i = 0; i < 16; ++i) {
      int K = (i < 8) ? (8 * hh + i) : (16 + 8 * hh + (i - 8));
      bg[i] = ldsw[K * 16 + mm];
    }

    // GEMM2: x'^T(k' x 16 nc) += (proj^T * ACT) @ tanh(grid)^T  (accumulate over 256 chunks)
    const v16h a20 = *(const v16h*)(projbank + (((size_t)c * 2 + 0) * 32 + lane) * 16);
    const v16h a21 = *(const v16h*)(projbank + (((size_t)c * 2 + 1) * 32 + lane) * 16);
    acc0 = __builtin_amdgcn_wmma_f32_16x16x32_f16(false, a20, false, bg, (short)0, acc0, false, false);
    acc1 = __builtin_amdgcn_wmma_f32_16x16x32_f16(false, a21, false, bg, (short)0, acc1, false, false);
  }

  // D layout: vgpr r, lane -> (k' = r + 8*hh [+16 for tile1], nc = mm)
#pragma unroll
  for (int r = 0; r < 8; ++r) {
    int kp = r + 8 * hh;
    float* orow = xprime + (size_t)(rowbase + mm) * 32;
    orow[kp]      = acc0[r];
    orow[16 + kp] = acc1[r];
  }
}

// ---------------- epilogue: out[n,j,k] = sum_i x'[n,i,k] * wl[j,i,l(k)] ----------------
__global__ void out_kernel(const float* __restrict__ xprime, const float* __restrict__ wl,
                           float* __restrict__ out) {
  int tid = blockIdx.x * blockDim.x + threadIdx.x;    // 1024*32*25 = 819200
  if (tid >= NNODES * COUT * KSH) return;
  int k = tid % KSH;
  int j = (tid / KSH) % COUT;
  int n = tid / (KSH * COUT);
  int l = c_lofk[k];
  float acc = 0.0f;
#pragma unroll 8
  for (int i = 0; i < CIN; ++i)
    acc += xprime[(size_t)(n * CIN + i) * 32 + k] * wl[(j * CIN + i) * 5 + l];
  out[tid] = acc;
}

extern "C" void kernel_launch(void* const* d_in, const int* in_sizes, int n_in,
                              void* d_out, int out_size, void* d_ws, size_t ws_size,
                              hipStream_t stream) {
  (void)in_sizes; (void)n_in; (void)out_size; (void)ws_size;
  const float* x = (const float*)d_in[0];             // (1024, 32, 25) f32
  const float* h = (const float*)d_in[1];             // (32, 32, 64) f32
  float* out = (float*)d_out;                          // (1024, 32, 25) f32
  char* ws = (char*)d_ws;

  float*     ynode    = (float*)(ws + WS_Y);
  float*     qw       = (float*)(ws + WS_QW);
  float*     Pn       = (float*)(ws + WS_PN);
  _Float16*  bank1    = (_Float16*)(ws + WS_BANK1);
  _Float16*  projbank = (_Float16*)(ws + WS_PROJ);
  float*     wl       = (float*)(ws + WS_WL);
  float*     xprime   = (float*)(ws + WS_XP);

  gl_setup_kernel<<<1, 64, 0, stream>>>(ynode, qw, Pn);
  fill_bank1_kernel<<<1024, 256, 0, stream>>>(Pn, bank1);
  fill_proj_kernel<<<1024, 256, 0, stream>>>(Pn, qw, projbank);
  fill_w_kernel<<<20, 256, 0, stream>>>(h, Pn, qw, wl);
  sph_main_kernel<<<NTILES / 8, 256, 0, stream>>>(x, bank1, projbank, xprime);
  out_kernel<<<3200, 256, 0, stream>>>(xprime, wl, out);
}
